// TripletLoss_37812892074050
// MI455X (gfx1250) — matrix operand order repaired and checked by reading, not compiled, not run on hardware
//
#include <hip/hip_runtime.h>
#include <float.h>

// ---------------- types ----------------
typedef __attribute__((ext_vector_type(16))) __bf16 v16bf;
typedef __attribute__((ext_vector_type(8)))  __bf16 v8bf;
typedef __attribute__((ext_vector_type(4)))  __bf16 v4bf;
typedef __attribute__((ext_vector_type(8)))  float  v8f;
typedef __attribute__((ext_vector_type(4)))  unsigned int u32x4;
typedef __attribute__((ext_vector_type(8)))  int i32x8;
typedef __attribute__((ext_vector_type(4)))  int i32x4;

#define NROWS   4096
#define DIM     512
#define MARGIN  200.0f
#define KC      32                  // K-chunk per TDM tile
#define LDK     (KC + 8)            // padded LDS stride: 40 bf16 = 80B (16B aligned)
#define ROWTILE 128
#define COLGROUPS 8
#define COLTILES_PER_BLOCK (32 / COLGROUPS)
#define NKC     (DIM / KC)          // 16 K-chunks
#define TILE_ELEMS (ROWTILE * LDK)  // 5120 bf16
#define TILE_BYTES (TILE_ELEMS * 2) // 10240 B

// monotonic uint encoding of float (order-preserving) for atomic max/min
__device__ __forceinline__ unsigned f2ord(float f) {
  unsigned u = __float_as_uint(f);
  return (u & 0x80000000u) ? ~u : (u | 0x80000000u);
}
__device__ __forceinline__ float ord2f(unsigned u) {
  return __uint_as_float((u & 0x80000000u) ? (u & 0x7fffffffu) : ~u);
}

// 16-elem bf16 WMMA fragment = two aligned 16B LDS chunks
__device__ __forceinline__ v16bf frag16(const __bf16* base, int off0, int off1) {
  v8bf x = *(const v8bf*)(base + off0);
  v8bf y = *(const v8bf*)(base + off1);
  return __builtin_shufflevector(x, y, 0,1,2,3,4,5,6,7,8,9,10,11,12,13,14,15);
}

__device__ __forceinline__ void split4(float4 v, v4bf& hi, v4bf& lo) {
  __bf16 h0 = (__bf16)v.x, h1 = (__bf16)v.y, h2 = (__bf16)v.z, h3 = (__bf16)v.w;
  __bf16 l0 = (__bf16)(v.x - (float)h0);
  __bf16 l1 = (__bf16)(v.y - (float)h1);
  __bf16 l2 = (__bf16)(v.z - (float)h2);
  __bf16 l3 = (__bf16)(v.w - (float)h3);
  hi[0]=h0; hi[1]=h1; hi[2]=h2; hi[3]=h3;
  lo[0]=l0; lo[1]=l1; lo[2]=l2; lo[3]=l3;
}

// ---- TDM: DMA one 128x32 bf16 tile (row stride 512 elems) into LDS with
// pad_interval=16 dwords (64B row) / pad_amount=4 dwords (16B) -> LDS stride 80B = LDK.
__device__ __forceinline__ void tdm_load_tile(unsigned ldsOff, const __bf16* gptr) {
  unsigned long long ga = (unsigned long long)(uintptr_t)gptr;
  u32x4 g0;
  g0[0] = 1u;                                   // count=1, user mode, no gather
  g0[1] = ldsOff;                               // lds_addr (bytes)
  g0[2] = (unsigned)ga;                         // global_addr[31:0]
  g0[3] = (unsigned)(ga >> 32) | (2u << 30);    // global_addr[56:32] | type=2
  i32x8 g1;
  g1[0] = (1 << 16)      // data_size = 2 bytes
        | (1 << 20)      // pad_enable
        | (3 << 22)      // pad_interval: 16 dwords stored ...
        | (3 << 25);     // pad_amount:   ... then skip 4 dwords
  g1[1] = (int)(512u << 16);    // tensor_dim0[15:0] = 512
  g1[2] = (int)(4096u << 16);   // tensor_dim0[31:16]=0 | tensor_dim1[15:0] = 4096
  g1[3] = (int)(32u << 16);     // tensor_dim1[31:16]=0 | tile_dim0 = 32
  g1[4] = 128;                  // tile_dim1 = 128, tile_dim2 = 0
  g1[5] = 512;                  // tensor_dim0_stride[31:0] = 512
  g1[6] = 0;                    // stride hi | tensor_dim1_stride lo (unused, 2D)
  g1[7] = 0;
  i32x4 z4 = {0, 0, 0, 0};
  i32x8 z8 = {0, 0, 0, 0, 0, 0, 0, 0};
  // 6-arg form (clang-23 / therock signature): (g0, g1, g2, g3, extra, cpol)
  __builtin_amdgcn_tensor_load_to_lds(g0, g1, z4, z4, z8, 0);
}

__device__ __forceinline__ void issue_chunk(unsigned ldsBase, int buf,
                                            const __bf16* Xhi, const __bf16* Xlo,
                                            int rowBase, int colBase, int kc) {
  unsigned o = ldsBase + (unsigned)(buf * 4 * TILE_BYTES);
  const int ra = rowBase * DIM + kc;
  const int rb = colBase * DIM + kc;
  tdm_load_tile(o + 0 * TILE_BYTES, Xhi + ra);
  tdm_load_tile(o + 1 * TILE_BYTES, Xlo + ra);
  tdm_load_tile(o + 2 * TILE_BYTES, Xhi + rb);
  tdm_load_tile(o + 3 * TILE_BYTES, Xlo + rb);
}

// ---------------- kernel 1: split X -> (hi,lo) bf16 planes, row norms, inits ----------------
__global__ __launch_bounds__(256)
void prep_kernel(const float* __restrict__ X, float* __restrict__ sq,
                 unsigned* __restrict__ apb, unsigned* __restrict__ anb,
                 __bf16* __restrict__ Xhi, __bf16* __restrict__ Xlo) {
  int wave = threadIdx.x >> 5;
  int lane = threadIdx.x & 31;
  int row  = blockIdx.x * 8 + wave;
  const float4* p = (const float4*)(X + (size_t)row * DIM);
  float s = 0.f;
  #pragma unroll
  for (int i = 0; i < 4; ++i) {
    int idx4 = lane + i * 32;                 // float4 index in row
    if (i < 3) __builtin_prefetch(p + idx4 + 32, 0, 3);
    float4 v = p[idx4];
    s += v.x * v.x + v.y * v.y + v.z * v.z + v.w * v.w;
    v4bf hi, lo;
    split4(v, hi, lo);
    *(v4bf*)(Xhi + (size_t)row * DIM + idx4 * 4) = hi;
    *(v4bf*)(Xlo + (size_t)row * DIM + idx4 * 4) = lo;
  }
  #pragma unroll
  for (int off = 16; off >= 1; off >>= 1) s += __shfl_xor(s, off, 32);
  if (lane == 0) {
    sq[row]  = s;
    apb[row] = 0u;           // encoded -inf for running max
    anb[row] = 0xFFFFFFFFu;  // encoded +inf for running min
  }
}

// ---------------- kernel 2: TDM-fed WMMA Gram + masked hard mining ----------------
__global__ __launch_bounds__(256)
void triplet_tiles(const __bf16* __restrict__ Xhi, const __bf16* __restrict__ Xlo,
                   const int* __restrict__ tgt, const float* __restrict__ sq,
                   unsigned* __restrict__ apb, unsigned* __restrict__ anb) {
  // [2 buffers][Ah, Al, Bh, Bl][128 x 40 bf16]  = 80 KB
  __shared__ __bf16 smem[2 * 4 * TILE_ELEMS];

  const int rowBase = blockIdx.x * ROWTILE;
  const int cg   = blockIdx.y;
  const int wave = threadIdx.x >> 5;
  const int lane = threadIdx.x & 31;
  const int h    = lane >> 4;
  const int r16  = lane & 15;
  const unsigned ldsBase = (unsigned)(uintptr_t)(void*)smem;

  // per-row constants for this wave's strip (C/D row = v + 8*h)
  float rs[8]; int tr[8];
  #pragma unroll
  for (int v = 0; v < 8; ++v) {
    int row = rowBase + wave * 16 + v + 8 * h;
    rs[v] = sq[row];
    tr[v] = tgt[row];
  }

  float ap[8], an[8];
  #pragma unroll
  for (int v = 0; v < 8; ++v) { ap[v] = -FLT_MAX; an[v] = FLT_MAX; }

  const v8f vzero = {0.f, 0.f, 0.f, 0.f, 0.f, 0.f, 0.f, 0.f};

  for (int t = 0; t < COLTILES_PER_BLOCK; ++t) {
    const int colBase = (cg * COLTILES_PER_BLOCK + t) * ROWTILE;
    v8f acc[8];
    #pragma unroll
    for (int n = 0; n < 8; ++n) acc[n] = vzero;

    __syncthreads();  // everyone done with previous tile's LDS before restaging
    if (threadIdx.x < 32)
      issue_chunk(ldsBase, 0, Xhi, Xlo, rowBase, colBase, 0);

    for (int kci = 0; kci < NKC; ++kci) {
      if (threadIdx.x < 32) __builtin_amdgcn_s_wait_tensorcnt(0);
      __syncthreads();  // chunk kci resident; all waves done with chunk kci-1
      if ((kci + 1 < NKC) && (threadIdx.x < 32))
        issue_chunk(ldsBase, (kci + 1) & 1, Xhi, Xlo, rowBase, colBase, (kci + 1) * KC);

      const __bf16* buf = smem + (size_t)((kci & 1) * 4 * TILE_ELEMS);
      const __bf16* Ahp = buf + 0 * TILE_ELEMS + (wave * 16 + r16) * LDK;
      const __bf16* Alp = buf + 1 * TILE_ELEMS + (wave * 16 + r16) * LDK;
      v16bf afh = frag16(Ahp, 8 * h, 16 + 8 * h);
      v16bf afl = frag16(Alp, 8 * h, 16 + 8 * h);
      #pragma unroll
      for (int n = 0; n < 8; ++n) {
        const __bf16* Bhp = buf + 2 * TILE_ELEMS + (n * 16 + r16) * LDK;
        const __bf16* Blp = buf + 3 * TILE_ELEMS + (n * 16 + r16) * LDK;
        v16bf bfh = frag16(Bhp, 8 * h, 16 + 8 * h);
        v16bf bfl = frag16(Blp, 8 * h, 16 + 8 * h);
        // x.y ~= hi*hi + hi*lo + lo*hi   (f32 accumulate)
        acc[n] = __builtin_amdgcn_wmma_f32_16x16x32_bf16(false, afl, false, bfh,
                                                         (short)0, acc[n], false, false);
        acc[n] = __builtin_amdgcn_wmma_f32_16x16x32_bf16(false, afh, false, bfl,
                                                         (short)0, acc[n], false, false);
        acc[n] = __builtin_amdgcn_wmma_f32_16x16x32_bf16(false, afh, false, bfh,
                                                         (short)0, acc[n], false, false);
      }
    }

    // epilogue: dist = sq_i + sq_j - 2*G; masked running max/min (registers only)
    #pragma unroll
    for (int n = 0; n < 8; ++n) {
      int col = colBase + n * 16 + r16;
      float cs = sq[col];
      int   tc = tgt[col];
      #pragma unroll
      for (int v = 0; v < 8; ++v) {
        float dval = rs[v] + cs - 2.0f * acc[n][v];
        if (tr[v] == tc) ap[v] = fmaxf(ap[v], dval);
        else             an[v] = fminf(an[v], dval);
      }
    }
  }

  // reduce across the 16 lanes of each half-wave (xor<=8 stays within a half)
  #pragma unroll
  for (int v = 0; v < 8; ++v) {
    float a = ap[v], b = an[v];
    #pragma unroll
    for (int off = 8; off >= 1; off >>= 1) {
      a = fmaxf(a, __shfl_xor(a, off, 32));
      b = fminf(b, __shfl_xor(b, off, 32));
    }
    if (r16 == 0) {
      int row = rowBase + wave * 16 + v + 8 * h;
      atomicMax(&apb[row], f2ord(a));
      atomicMin(&anb[row], f2ord(b));
    }
  }
}

// ---------------- kernel 3: mean(relu(ap - an + margin)) ----------------
__global__ __launch_bounds__(256)
void finalize_kernel(const unsigned* __restrict__ apb, const unsigned* __restrict__ anb,
                     float* __restrict__ out) {
  __shared__ float red[8];
  float s = 0.f;
  for (int i = threadIdx.x; i < NROWS; i += 256) {
    float a = ord2f(apb[i]);
    float b = ord2f(anb[i]);
    s += fmaxf(0.f, a - b + MARGIN);
  }
  #pragma unroll
  for (int off = 16; off >= 1; off >>= 1) s += __shfl_xor(s, off, 32);
  if ((threadIdx.x & 31) == 0) red[threadIdx.x >> 5] = s;
  __syncthreads();
  if (threadIdx.x == 0) {
    float tsum = 0.f;
    #pragma unroll
    for (int i = 0; i < 8; ++i) tsum += red[i];
    out[0] = tsum * (1.0f / (float)NROWS);
  }
}

// ---------------- launcher ----------------
extern "C" void kernel_launch(void* const* d_in, const int* in_sizes, int n_in,
                              void* d_out, int out_size, void* d_ws, size_t ws_size,
                              hipStream_t stream) {
  const float* X  = (const float*)d_in[0];
  const int* tgt  = (const int*)d_in[1];
  float*    sq    = (float*)d_ws;                              // [4096] f32
  unsigned* apb   = (unsigned*)d_ws + NROWS;                   // [4096] u32
  unsigned* anb   = (unsigned*)d_ws + 2 * NROWS;               // [4096] u32
  __bf16*   Xhi   = (__bf16*)((char*)d_ws + 3 * NROWS * 4);    // [4096*512] bf16
  __bf16*   Xlo   = Xhi + (size_t)NROWS * DIM;                 // [4096*512] bf16
  float*    out   = (float*)d_out;

  prep_kernel<<<dim3(NROWS / 8), dim3(256), 0, stream>>>(X, sq, apb, anb, Xhi, Xlo);
  triplet_tiles<<<dim3(NROWS / ROWTILE, COLGROUPS), dim3(256), 0, stream>>>(Xhi, Xlo, tgt, sq, apb, anb);
  finalize_kernel<<<dim3(1), dim3(256), 0, stream>>>(apb, anb, out);
}
